// Attention_89962384982638
// MI455X (gfx1250) — matrix operand order repaired and checked by reading, not compiled, not run on hardware
//
#include <hip/hip_runtime.h>
#include <hip/hip_bf16.h>

#define B_ 4
#define H_ 8
#define S_ 2048
#define D_ 64
#define CHUNK 32
#define WAVES 4

typedef __attribute__((ext_vector_type(16))) __bf16 v16bf;
typedef __attribute__((ext_vector_type(8)))  float  v8f;

union Frag16 {
  v16bf v;
  unsigned short s[16];
  unsigned int   u[8];
  uint4          q[2];
};

__device__ __forceinline__ unsigned short f2bf(float x) {
  unsigned int u = __builtin_bit_cast(unsigned int, x);
  u += 0x7fffu + ((u >> 16) & 1u);          // round-to-nearest-even
  return (unsigned short)(u >> 16);
}
__device__ __forceinline__ unsigned int pack2(unsigned short lo, unsigned short hi) {
  return (unsigned int)lo | ((unsigned int)hi << 16);
}
__device__ __forceinline__ float fast_exp2(float x) {
#if __has_builtin(__builtin_amdgcn_exp2f)
  return __builtin_amdgcn_exp2f(x);         // v_exp_f32
#else
  return exp2f(x);
#endif
}
__device__ __forceinline__ float fast_rcp(float x) {
#if __has_builtin(__builtin_amdgcn_rcpf)
  return __builtin_amdgcn_rcpf(x);          // v_rcp_f32
#else
  return 1.0f / x;
#endif
}

// ---------------- pre-pass 1: f32 -> bf16 (pairs), optional scale ----------------
__global__ __launch_bounds__(256)
void cvt_bf16_scale(const float* __restrict__ src, unsigned int* __restrict__ dst,
                    float scale, int n2) {
  int i = blockIdx.x * blockDim.x + threadIdx.x;
  if (i < n2) {
    float2 v = ((const float2*)src)[i];
    dst[i] = pack2(f2bf(v.x * scale), f2bf(v.y * scale));
  }
}

// ---------------- pre-pass 2: V[b,h,s,d] f32 -> VT[b,h,d,s] bf16 ----------------
__global__ __launch_bounds__(256)
void vT_bf16(const float* __restrict__ V, unsigned int* __restrict__ VT) {
  __shared__ unsigned short T[64][72];           // padded: conflict-free
  const int bh = blockIdx.x >> 5;                // b*H + h
  const int s0 = (blockIdx.x & 31) * 64;
  const int t  = threadIdx.x;
  const float* vp = V + ((size_t)bh * S_ + s0) * D_;
#pragma unroll
  for (int i = 0; i < 16; ++i) {                 // coalesced reads
    int li = t + 256 * i;                        // 0..4095
    int d = li & 63, s = li >> 6;
    T[d][s] = f2bf(vp[(size_t)s * D_ + d]);
  }
  __syncthreads();
  unsigned int* out = VT + (size_t)bh * D_ * (S_ / 2);
#pragma unroll
  for (int i = 0; i < 8; ++i) {                  // coalesced packed writes
    int li = t + 256 * i;                        // 0..2047
    int sp = li & 31, d = li >> 5;
    out[(size_t)d * (S_ / 2) + (s0 >> 1) + sp] = pack2(T[d][2 * sp], T[d][2 * sp + 1]);
  }
}

// ---------------- main: batch-axis-softmax attention, bf16 WMMA ----------------
// One WG = (head, 64 i-rows); one wave = 16 i-rows x all 4 batches.
__global__ __launch_bounds__(128, 1)
void attn_bsoftmax_wmma(const unsigned int* __restrict__ Qbf,   // prescaled bf16 pairs
                        const unsigned int* __restrict__ Kbf,   // bf16 pairs
                        const unsigned int* __restrict__ VTbf,  // transposed bf16 pairs
                        float* __restrict__ Og) {
  // padded LDS, 16B-aligned rows, bank-friendly strides
  __shared__ unsigned int KshDW[B_ * CHUNK * 36]; // row = 32 dw data + 4 pad
  __shared__ unsigned int VshDW[B_ * D_ * 20];    // row = 16 dw data + 4 pad
  __shared__ unsigned int WshDW[WAVES * 16 * 20]; // per-wave 16x32 bf16 W tile

  const int tid  = threadIdx.x;
  const int wave = tid >> 5;
  const int lane = tid & 31;
  const int lrow = lane & 15;
  const int g    = lane >> 4;

  const int h  = blockIdx.x >> 5;
  const int it = blockIdx.x & 31;
  const int i0 = it * 64 + wave * 16;

  // ---- Q A-fragments (already bf16 + prescaled by log2(e)/sqrt(H)) ----
  Frag16 qf[B_][2];
#pragma unroll
  for (int b = 0; b < B_; ++b) {
    const uint4* qd = (const uint4*)(Qbf + ((size_t)(b * H_ + h) * S_ + i0 + lrow) * 32);
#pragma unroll
    for (int f = 0; f < 2; ++f) {
      qf[b][f].q[0] = qd[f * 4 + g];       // k = f*32 + g*8 + 0..7
      qf[b][f].q[1] = qd[f * 4 + 2 + g];   // k = f*32 + 16 + g*8 + 0..7
    }
  }

  v8f acc[B_][4];
#pragma unroll
  for (int b = 0; b < B_; ++b)
#pragma unroll
    for (int nt = 0; nt < 4; ++nt)
      acc[b][nt] = (v8f){0.f, 0.f, 0.f, 0.f, 0.f, 0.f, 0.f, 0.f};

  for (int jc = 0; jc < S_; jc += CHUNK) {
    __syncthreads();

    // ---- staging: pure b128 copies (bf16 already), padded destinations ----
#pragma unroll
    for (int i = 0; i < 8; ++i) {                      // K: 1024 granules / 128 thr
      int li = tid + 128 * i;
      int gr = li & 7, j = (li >> 3) & 31, b = (li >> 8) & 3;
      uint4 val = *(const uint4*)(Kbf + ((size_t)(b * H_ + h) * S_ + jc + j) * 32 + gr * 4);
      *(uint4*)&KshDW[b * 1152 + j * 36 + gr * 4] = val;
    }
#pragma unroll
    for (int i = 0; i < 8; ++i) {                      // V: 1024 granules / 128 thr
      int li = tid + 128 * i;
      int gr = li & 3, d = (li >> 2) & 63, b = (li >> 8) & 3;
      uint4 val = *(const uint4*)(VTbf + ((size_t)(b * H_ + h) * D_ + d) * 1024 + (jc >> 1) + gr * 4);
      *(uint4*)&VshDW[b * 1280 + d * 20 + gr * 4] = val;
    }
    if (jc + CHUNK < S_) {                             // global_prefetch_b8
      __builtin_prefetch(Kbf + ((size_t)(wave * H_ + h) * S_ + jc + CHUNK + lane) * 32, 0, 0);
      __builtin_prefetch(VTbf + ((size_t)(wave * H_ + h) * D_ + lane * 2) * 1024 + ((jc + CHUNK) >> 1), 0, 0);
    }
    __syncthreads();

    // ---- scores (log2-domain): S = Qpre . K^T ----
    v8f sc[B_][2];
#pragma unroll
    for (int b = 0; b < B_; ++b)
#pragma unroll
      for (int t = 0; t < 2; ++t) {
        v8f c = (v8f){0.f, 0.f, 0.f, 0.f, 0.f, 0.f, 0.f, 0.f};
#pragma unroll
        for (int d2 = 0; d2 < 2; ++d2) {
          Frag16 kb;
          const uint4* kp = (const uint4*)&KshDW[b * 1152 + (t * 16 + lrow) * 36 + d2 * 16 + g * 4];
          kb.q[0] = kp[0];
          kb.q[1] = kp[2];
          c = __builtin_amdgcn_wmma_f32_16x16x32_bf16(false, qf[b][d2].v,
                                                      false, kb.v,
                                                      (short)0, c, false, false);
        }
        sc[b][t] = c;
      }

    // ---- softmax over batch axis: w[b] = exp2(s[b]) / sum_b exp2(s[b]) ----
#pragma unroll
    for (int t = 0; t < 2; ++t)
#pragma unroll
      for (int k = 0; k < 8; ++k) {
        float e0 = fast_exp2(sc[0][t][k]);
        float e1 = fast_exp2(sc[1][t][k]);
        float e2 = fast_exp2(sc[2][t][k]);
        float e3 = fast_exp2(sc[3][t][k]);
        float r = fast_rcp(e0 + e1 + e2 + e3);
        sc[0][t][k] = e0 * r;
        sc[1][t][k] = e1 * r;
        sc[2][t][k] = e2 * r;
        sc[3][t][k] = e3 * r;
      }

    // ---- PV: O[b] += W[b](16x32) . V[b](32x64) ----
#pragma unroll
    for (int b = 0; b < B_; ++b) {
      // D-layout -> A-layout via per-wave LDS patch; bf16 by f32 truncation
      unsigned short* ws = (unsigned short*)&WshDW[wave * 320];
#pragma unroll
      for (int t = 0; t < 2; ++t)
#pragma unroll
        for (int r = 0; r < 8; ++r)
          ws[(r + 8 * g) * 40 + t * 16 + lrow] =
              (unsigned short)(__builtin_bit_cast(unsigned int, sc[b][t][r]) >> 16);
      Frag16 wa;
      const uint4* wp = (const uint4*)&WshDW[wave * 320 + lrow * 20];
      wa.q[0] = wp[g];       // k = g*8 + 0..7
      wa.q[1] = wp[2 + g];   // k = 16 + g*8 + 0..7

#pragma unroll
      for (int nt = 0; nt < 4; ++nt) {
        Frag16 vb;
        const uint4* vp4 = (const uint4*)&VshDW[b * 1280 + (nt * 16 + lrow) * 20 + g * 4];
        vb.q[0] = vp4[0];
        vb.q[1] = vp4[2];
        acc[b][nt] = __builtin_amdgcn_wmma_f32_16x16x32_bf16(false, wa.v,
                                                             false, vb.v,
                                                             (short)0, acc[b][nt],
                                                             false, false);
      }
    }
  }

  // ---- store O (f32) ----
#pragma unroll
  for (int b = 0; b < B_; ++b)
#pragma unroll
    for (int nt = 0; nt < 4; ++nt)
#pragma unroll
      for (int r = 0; r < 8; ++r)
        Og[((size_t)(b * H_ + h) * S_ + i0 + r + 8 * g) * D_ + nt * 16 + lrow] =
            acc[b][nt][r];
}

extern "C" void kernel_launch(void* const* d_in, const int* in_sizes, int n_in,
                              void* d_out, int out_size, void* d_ws, size_t ws_size,
                              hipStream_t stream) {
  (void)in_sizes; (void)n_in; (void)out_size; (void)ws_size;
  const float* Q = (const float*)d_in[0];
  const float* K = (const float*)d_in[1];
  const float* V = (const float*)d_in[2];
  float* O = (float*)d_out;

  const int nElem = B_ * H_ * S_ * D_;      // 4,194,304
  const int n2 = nElem / 2;                 // packed bf16 pairs
  unsigned int* Qbf  = (unsigned int*)d_ws;                 // 8 MB
  unsigned int* Kbf  = Qbf + n2;                            // 8 MB
  unsigned int* VTbf = Kbf + n2;                            // 8 MB  (24 MB total)

  // Q prescale: log2(e)/sqrt(H) -> scores land in exp2 domain
  const float qprescale = 1.4426950408889634f / 2.8284271247461903f;

  cvt_bf16_scale<<<dim3(n2 / 256), dim3(256), 0, stream>>>(Q, Qbf, qprescale, n2);
  cvt_bf16_scale<<<dim3(n2 / 256), dim3(256), 0, stream>>>(K, Kbf, 1.0f, n2);
  vT_bf16<<<dim3(B_ * H_ * (S_ / 64)), dim3(256), 0, stream>>>(V, VTbf);

  attn_bsoftmax_wmma<<<dim3(H_ * (S_ / 64)), dim3(128), 0, stream>>>(Qbf, Kbf, VTbf, O);
}
